// TransformerBlock_20349555048887
// MI455X (gfx1250) — matrix operand order repaired and checked by reading, not compiled, not run on hardware
//
#include <hip/hip_runtime.h>
#include <hip/hip_bf16.h>
#include <math.h>

typedef __attribute__((ext_vector_type(16))) _Float16 v16h;
typedef __attribute__((ext_vector_type(8)))  float    v8f;

#define D_MODEL 768
#define T_SEQ   4096
#define N_HEADS 12
#define D_HEAD  64
#define D_FF    3072

// Async global->LDS copy of 16 bytes (CDNA5, ASYNCcnt-tracked, bypasses VGPRs)
__device__ __forceinline__ void async_cp16(const void* lds_dst, const void* gsrc) {
  unsigned loff = (unsigned)(size_t)lds_dst;  // addrspace(3) offset
  unsigned long long ga = (unsigned long long)(size_t)gsrc;
  asm volatile("global_load_async_to_lds_b128 %0, %1, off"
               :: "v"(loff), "v"(ga) : "memory");
}
__device__ __forceinline__ void wait_async0() {
  asm volatile("s_wait_asynccnt 0" ::: "memory");
}

// ---------------------------------------------------------------------------
// f32 -> f16 cast (weights, done once per call)
// ---------------------------------------------------------------------------
__global__ void cast_f32_h16(const float* __restrict__ in,
                             _Float16* __restrict__ out, int n) {
  int i = blockIdx.x * blockDim.x + threadIdx.x;
  int stride = gridDim.x * blockDim.x;
  for (; i < n; i += stride) out[i] = (_Float16)in[i];
}

// ---------------------------------------------------------------------------
// LayerNorm: fp32 [T,768] -> f16 [T,768]; one row per 256-thread block
// ---------------------------------------------------------------------------
__global__ __launch_bounds__(256)
void layernorm_h16(const float* __restrict__ x, const float* __restrict__ gamma,
                   const float* __restrict__ beta, _Float16* __restrict__ out) {
  __shared__ float s_sum[256];
  __shared__ float s_sq[256];
  const int row = blockIdx.x;
  const int t = threadIdx.x;
  const float* xr = x + (size_t)row * D_MODEL;
  float v0 = xr[t], v1 = xr[t + 256], v2 = xr[t + 512];
  s_sum[t] = v0 + v1 + v2;
  s_sq[t]  = v0 * v0 + v1 * v1 + v2 * v2;
  __syncthreads();
  for (int s = 128; s > 0; s >>= 1) {
    if (t < s) { s_sum[t] += s_sum[t + s]; s_sq[t] += s_sq[t + s]; }
    __syncthreads();
  }
  const float mu  = s_sum[0] * (1.0f / D_MODEL);
  const float var = s_sq[0] * (1.0f / D_MODEL) - mu * mu;
  const float rs  = rsqrtf(var + 1e-5f);
  _Float16* orow = out + (size_t)row * D_MODEL;
  orow[t]       = (_Float16)((v0 - mu) * rs * gamma[t]       + beta[t]);
  orow[t + 256] = (_Float16)((v1 - mu) * rs * gamma[t + 256] + beta[t + 256]);
  orow[t + 512] = (_Float16)((v2 - mu) * rs * gamma[t + 512] + beta[t + 512]);
}

// ---------------------------------------------------------------------------
// WMMA GEMM: C = act(A[M,K]h16 @ B[K,N]h16 + bias) (+resid); out fp32/f16
// 128x128x32 tiles, 256 thr = 8 waves as 4(M)x2(N); wave tile 32x64.
// Double-buffered LDS; A tile staged with GLOBAL_LOAD_ASYNC_TO_LDS_B128,
// B tile prefetched global->reg, transposed reg->LDS after compute.
// ACT: 0=none 1=exact GELU
// ---------------------------------------------------------------------------
template <int ACT>
__global__ __launch_bounds__(256)
void gemm_wmma(const _Float16* __restrict__ A, const _Float16* __restrict__ B,
               const float* __restrict__ bias, const float* __restrict__ resid,
               float* __restrict__ outF, _Float16* __restrict__ outH,
               int M, int N, int K) {
  __shared__ _Float16 As[2][128][40];  // [buf][m][k] padded
  __shared__ _Float16 Bs[2][128][40];  // [buf][n][k] (B staged transposed)

  const int tid = threadIdx.x;
  const int wave = tid >> 5, lane = tid & 31;
  const int g = lane >> 4, rc = lane & 15;
  const int wm = wave >> 1, wn = wave & 1;
  const int m0 = blockIdx.y * 128, n0 = blockIdx.x * 128;
  const int nk = K >> 5;

  v8f zero = {};
  v8f acc[2][4];
#pragma unroll
  for (int mi = 0; mi < 2; mi++)
#pragma unroll
    for (int ni = 0; ni < 4; ni++) acc[mi][ni] = zero;

  // ---- stage tile 0 into buffer 0 ----
#pragma unroll
  for (int it = 0; it < 2; it++) {          // A: 512 x 8-half chunks, async
    int ci = tid + it * 256;
    int row = ci >> 2, cc = ci & 3;
    async_cp16(&As[0][row][cc * 8], A + (size_t)(m0 + row) * K + cc * 8);
  }
#pragma unroll
  for (int it = 0; it < 2; it++) {          // B: load + transpose-store
    int ci = tid + it * 256;
    int kk = ci >> 4, nc = ci & 15;
    uint4 d = *(const uint4*)(B + (size_t)kk * N + n0 + nc * 8);
    _Float16 tmp[8];
    __builtin_memcpy(tmp, &d, 16);
#pragma unroll
    for (int e = 0; e < 8; e++) Bs[0][nc * 8 + e][kk] = tmp[e];
  }
  wait_async0();
  __syncthreads();

  int cur = 0;
  for (int i = 0; i < nk; ++i) {
    const int nxt = cur ^ 1;
    const bool pf = (i + 1) < nk;
    uint4 bpre[2];
    if (pf) {
      const int kn = (i + 1) << 5;
#pragma unroll
      for (int it = 0; it < 2; it++) {      // async A(i+1) -> As[nxt]
        int ci = tid + it * 256;
        int row = ci >> 2, cc = ci & 3;
        async_cp16(&As[nxt][row][cc * 8],
                   A + (size_t)(m0 + row) * K + kn + cc * 8);
      }
#pragma unroll
      for (int it = 0; it < 2; it++) {      // B(i+1) global -> regs
        int ci = tid + it * 256;
        int kk = ci >> 4, nc = ci & 15;
        bpre[it] = *(const uint4*)(B + (size_t)(kn + kk) * N + n0 + nc * 8);
      }
    }

    // ---- compute on buffer `cur` ----
    v16h afr[2], bfr[4];
#pragma unroll
    for (int mi = 0; mi < 2; mi++) {
      const _Float16* ap = &As[cur][wm * 32 + mi * 16 + rc][0];
#pragma unroll
      for (int j = 0; j < 8; j++) {
        afr[mi][j]     = ap[8 * g + j];
        afr[mi][8 + j] = ap[16 + 8 * g + j];
      }
    }
#pragma unroll
    for (int ni = 0; ni < 4; ni++) {
      const _Float16* bp = &Bs[cur][wn * 64 + ni * 16 + rc][0];
#pragma unroll
      for (int j = 0; j < 16; j++) bfr[ni][j] = bp[(j < 8 ? 8 * g + j : 8 + 8 * g + j)];
    }
#pragma unroll
    for (int mi = 0; mi < 2; mi++)
#pragma unroll
      for (int ni = 0; ni < 4; ni++)
        acc[mi][ni] = __builtin_amdgcn_wmma_f32_16x16x32_f16(
            false, afr[mi], false, bfr[ni], (short)0, acc[mi][ni], false, false);

    if (pf) {
#pragma unroll
      for (int it = 0; it < 2; it++) {      // B(i+1) regs -> Bs[nxt] transposed
        int ci = tid + it * 256;
        int kk = ci >> 4, nc = ci & 15;
        _Float16 tmp[8];
        __builtin_memcpy(tmp, &bpre[it], 16);
#pragma unroll
        for (int e = 0; e < 8; e++) Bs[nxt][nc * 8 + e][kk] = tmp[e];
      }
      wait_async0();                        // A(i+1) landed in LDS
    }
    __syncthreads();                        // tile i+1 visible block-wide
    cur = nxt;
  }

  // Epilogue: C-frag layout row = 8g+i, col = rc
#pragma unroll
  for (int mi = 0; mi < 2; mi++) {
#pragma unroll
    for (int ni = 0; ni < 4; ni++) {
      const int col = n0 + wn * 64 + ni * 16 + rc;
      const float bv = bias ? bias[col] : 0.0f;
#pragma unroll
      for (int i = 0; i < 8; i++) {
        const int row = m0 + wm * 32 + mi * 16 + 8 * g + i;
        float v = acc[mi][ni][i] + bv;
        if (ACT == 1) v = 0.5f * v * (1.0f + erff(v * 0.70710678f));
        if (resid) v += resid[(size_t)row * N + col];
        if (outF) outF[(size_t)row * N + col] = v;
        if (outH) outH[(size_t)row * N + col] = (_Float16)v;
      }
    }
  }
}

// ---------------------------------------------------------------------------
// Flash attention (causal), per (64-row q tile, head). 128 thr = 4 waves,
// each wave owns 16 query rows. Online softmax; WMMA for QK^T and PV.
// Q/K tiles staged with async-to-LDS; V transposed through registers.
// ---------------------------------------------------------------------------
__global__ __launch_bounds__(128)
void attn_wmma(const _Float16* __restrict__ Q, const _Float16* __restrict__ Kg,
               const _Float16* __restrict__ Vg, _Float16* __restrict__ Og) {
  __shared__ _Float16 Qs[64][72];       // [q][dh]
  __shared__ _Float16 Ks[32][72];       // [s][dh]
  __shared__ _Float16 VsT[64][40];      // [dh][s]  (transposed for B-frag)
  __shared__ _Float16 Ps[4][16][40];    // wave-private 16x32 P staging

  const int tid = threadIdx.x;
  const int wave = tid >> 5, lane = tid & 31;
  const int g = lane >> 4, rc = lane & 15;
  const int qbase = blockIdx.x * 64;
  const int col0 = blockIdx.y * D_HEAD;
  const int wr = wave * 16;

  // Async-load 64x64 Q tile
#pragma unroll
  for (int it = 0; it < 4; it++) {
    int ci = tid + it * 128;
    int row = ci >> 3, dc = ci & 7;
    async_cp16(&Qs[row][dc * 8],
               Q + (size_t)(qbase + row) * D_MODEL + col0 + dc * 8);
  }

  v8f zero = {};
  v8f o[4];
#pragma unroll
  for (int n = 0; n < 4; n++) o[n] = zero;
  float mrow[8], lrow[8];
#pragma unroll
  for (int i = 0; i < 8; i++) { mrow[i] = -1e30f; lrow[i] = 0.0f; }

  const int nblocks = (qbase + 64) >> 5;  // causal: keys up to qbase+63
  for (int blk = 0; blk < nblocks; blk++) {
    const int sb = blk * 32;
    __syncthreads();  // protect Ks/VsT reuse
#pragma unroll
    for (int it = 0; it < 2; it++) {
      int ci = tid + it * 128;
      int s = ci >> 3, dc = ci & 7;
      async_cp16(&Ks[s][dc * 8],
                 Kg + (size_t)(sb + s) * D_MODEL + col0 + dc * 8);
      uint4 d = *(const uint4*)(Vg + (size_t)(sb + s) * D_MODEL + col0 + dc * 8);
      _Float16 tmp[8];
      __builtin_memcpy(tmp, &d, 16);
#pragma unroll
      for (int e = 0; e < 8; e++) VsT[dc * 8 + e][s] = tmp[e];
    }
    wait_async0();       // per-wave asyncs done before signaling barrier
    __syncthreads();

    // S = Q(16x64) * K^T(64x32) over 2 k-steps, two N-tiles (cols 0-15,16-31)
    v8f s0 = zero, s1 = zero;
#pragma unroll
    for (int ks = 0; ks < 2; ks++) {
      const int kk = ks * 32;
      v16h a, b0, b1;
      const _Float16* qp = &Qs[wr + rc][kk];
#pragma unroll
      for (int j = 0; j < 8; j++) { a[j] = qp[8 * g + j]; a[8 + j] = qp[16 + 8 * g + j]; }
      const _Float16* k0p = &Ks[rc][kk];
      const _Float16* k1p = &Ks[16 + rc][kk];
#pragma unroll
      for (int j = 0; j < 16; j++) { b0[j] = k0p[(j < 8 ? 8 * g + j : 8 + 8 * g + j)]; b1[j] = k1p[(j < 8 ? 8 * g + j : 8 + 8 * g + j)]; }
      s0 = __builtin_amdgcn_wmma_f32_16x16x32_f16(false, a, false, b0, (short)0, s0, false, false);
      s1 = __builtin_amdgcn_wmma_f32_16x16x32_f16(false, a, false, b1, (short)0, s1, false, false);
    }

    // Online softmax: row stats live in lanes of each 16-lane column group
    float fac[8];
#pragma unroll
    for (int i = 0; i < 8; i++) {
      const int gr = qbase + wr + 8 * g + i;
      float x0 = s0[i] * 0.125f;  // 1/sqrt(64)
      float x1 = s1[i] * 0.125f;
      if (sb + rc > gr)      x0 = -1e30f;
      if (sb + 16 + rc > gr) x1 = -1e30f;
      float t = fmaxf(x0, x1);
      t = fmaxf(t, __shfl_xor(t, 1, 32));
      t = fmaxf(t, __shfl_xor(t, 2, 32));
      t = fmaxf(t, __shfl_xor(t, 4, 32));
      t = fmaxf(t, __shfl_xor(t, 8, 32));
      const float mn = fmaxf(mrow[i], t);
      const float f = expf(mrow[i] - mn);
      const float p0 = expf(x0 - mn);
      const float p1 = expf(x1 - mn);
      float rs = p0 + p1;
      rs += __shfl_xor(rs, 1, 32);
      rs += __shfl_xor(rs, 2, 32);
      rs += __shfl_xor(rs, 4, 32);
      rs += __shfl_xor(rs, 8, 32);
      mrow[i] = mn;
      lrow[i] = lrow[i] * f + rs;
      fac[i] = f;
      Ps[wave][8 * g + i][rc] = (_Float16)p0;       // C-frag -> LDS
      Ps[wave][8 * g + i][16 + rc] = (_Float16)p1;
    }
#pragma unroll
    for (int n = 0; n < 4; n++)
#pragma unroll
      for (int i = 0; i < 8; i++) o[n][i] *= fac[i];

    // Same-wave LDS RAW: wait for Ps stores before re-reading as A-fragment
    asm volatile("s_wait_dscnt 0" ::: "memory");

    v16h ap;
    const _Float16* pp = &Ps[wave][rc][0];
#pragma unroll
    for (int j = 0; j < 8; j++) { ap[j] = pp[8 * g + j]; ap[8 + j] = pp[16 + 8 * g + j]; }
#pragma unroll
    for (int n = 0; n < 4; n++) {
      v16h bv;
      const _Float16* vp = &VsT[n * 16 + rc][0];
#pragma unroll
      for (int j = 0; j < 16; j++) bv[j] = vp[(j < 8 ? 8 * g + j : 8 + 8 * g + j)];
      o[n] = __builtin_amdgcn_wmma_f32_16x16x32_f16(false, ap, false, bv, (short)0, o[n], false, false);
    }
  }

  // Normalize and write f16 output [T, 768]
#pragma unroll
  for (int i = 0; i < 8; i++) {
    const float inv = 1.0f / lrow[i];
    const int row = qbase + wr + 8 * g + i;
#pragma unroll
    for (int n = 0; n < 4; n++)
      Og[(size_t)row * D_MODEL + col0 + n * 16 + rc] = (_Float16)(o[n][i] * inv);
  }
}

// ---------------------------------------------------------------------------
// Orchestration
// ---------------------------------------------------------------------------
extern "C" void kernel_launch(void* const* d_in, const int* in_sizes, int n_in,
                              void* d_out, int out_size, void* d_ws, size_t ws_size,
                              hipStream_t stream) {
  (void)in_sizes; (void)n_in; (void)out_size; (void)ws_size;
  const float* x   = (const float*)d_in[0];
  const float* Wq  = (const float*)d_in[1];
  const float* Wk  = (const float*)d_in[2];
  const float* Wv  = (const float*)d_in[3];
  const float* Wo  = (const float*)d_in[4];
  const float* bo  = (const float*)d_in[5];
  const float* W1  = (const float*)d_in[6];
  const float* b1  = (const float*)d_in[7];
  const float* W2  = (const float*)d_in[8];
  const float* b2  = (const float*)d_in[9];
  const float* g1  = (const float*)d_in[10];
  const float* be1 = (const float*)d_in[11];
  const float* g2  = (const float*)d_in[12];
  const float* be2 = (const float*)d_in[13];
  float* out = (float*)d_out;

  char* ws = (char*)d_ws;
  size_t off = 0;
  auto alloc = [&](size_t bytes) -> char* {
    char* p = ws + off;
    off += (bytes + 255) & ~(size_t)255;
    return p;
  };

  const size_t TD = (size_t)T_SEQ * D_MODEL;
  _Float16* h1   = (_Float16*)alloc(TD * 2);
  _Float16* Wq16 = (_Float16*)alloc((size_t)D_MODEL * D_MODEL * 2);
  _Float16* Wk16 = (_Float16*)alloc((size_t)D_MODEL * D_MODEL * 2);
  _Float16* Wv16 = (_Float16*)alloc((size_t)D_MODEL * D_MODEL * 2);
  _Float16* Wo16 = (_Float16*)alloc((size_t)D_MODEL * D_MODEL * 2);
  _Float16* W116 = (_Float16*)alloc((size_t)D_MODEL * D_FF * 2);
  _Float16* W216 = (_Float16*)alloc((size_t)D_FF * D_MODEL * 2);
  _Float16* q16  = (_Float16*)alloc(TD * 2);
  _Float16* k16  = (_Float16*)alloc(TD * 2);
  _Float16* v16  = (_Float16*)alloc(TD * 2);
  _Float16* a16  = (_Float16*)alloc(TD * 2);
  float*    x1   = (float*)alloc(TD * 4);
  _Float16* h2   = (_Float16*)alloc(TD * 2);
  _Float16* f16b = (_Float16*)alloc((size_t)T_SEQ * D_FF * 2);

  // Cast weights to f16
  const int nDD = D_MODEL * D_MODEL;
  const int nDF = D_MODEL * D_FF;
  cast_f32_h16<<<(nDD + 255) / 256, 256, 0, stream>>>(Wq, Wq16, nDD);
  cast_f32_h16<<<(nDD + 255) / 256, 256, 0, stream>>>(Wk, Wk16, nDD);
  cast_f32_h16<<<(nDD + 255) / 256, 256, 0, stream>>>(Wv, Wv16, nDD);
  cast_f32_h16<<<(nDD + 255) / 256, 256, 0, stream>>>(Wo, Wo16, nDD);
  cast_f32_h16<<<(nDF + 255) / 256, 256, 0, stream>>>(W1, W116, nDF);
  cast_f32_h16<<<(nDF + 255) / 256, 256, 0, stream>>>(W2, W216, nDF);

  // h1 = LN(x)
  layernorm_h16<<<T_SEQ, 256, 0, stream>>>(x, g1, be1, h1);

  // Q, K, V projections
  dim3 gProj(D_MODEL / 128, T_SEQ / 128);
  gemm_wmma<0><<<gProj, 256, 0, stream>>>(h1, Wq16, nullptr, nullptr, nullptr, q16,
                                          T_SEQ, D_MODEL, D_MODEL);
  gemm_wmma<0><<<gProj, 256, 0, stream>>>(h1, Wk16, nullptr, nullptr, nullptr, k16,
                                          T_SEQ, D_MODEL, D_MODEL);
  gemm_wmma<0><<<gProj, 256, 0, stream>>>(h1, Wv16, nullptr, nullptr, nullptr, v16,
                                          T_SEQ, D_MODEL, D_MODEL);

  // Causal flash attention
  dim3 gAttn(T_SEQ / 64, N_HEADS);
  attn_wmma<<<gAttn, 128, 0, stream>>>(q16, k16, v16, a16);

  // x1 = x + attn @ Wo + bo
  gemm_wmma<0><<<gProj, 256, 0, stream>>>(a16, Wo16, bo, x, x1, nullptr,
                                          T_SEQ, D_MODEL, D_MODEL);

  // h2 = LN(x1)
  layernorm_h16<<<T_SEQ, 256, 0, stream>>>(x1, g2, be2, h2);

  // ff = gelu(h2 @ W1 + b1)
  dim3 gFF1(D_FF / 128, T_SEQ / 128);
  gemm_wmma<1><<<gFF1, 256, 0, stream>>>(h2, W116, b1, nullptr, nullptr, f16b,
                                         T_SEQ, D_FF, D_MODEL);

  // out = x1 + ff @ W2 + b2
  gemm_wmma<0><<<gProj, 256, 0, stream>>>(f16b, W216, b2, x1, out, nullptr,
                                          T_SEQ, D_MODEL, D_FF);
}
